// ModifiedMFF_36696200577359
// MI455X (gfx1250) — matrix operand order repaired and checked by reading, not compiled, not run on hardware
//
#include <hip/hip_runtime.h>
#include <cstdint>
#include <cstddef>

#define BATCH 4
#define DIMC  48
#define HIDC  96
#define NSET  32
#define HH    128
#define WWID  128
#define PHW   (HH*WWID)           // 16384
#define NPIX  (BATCH*PHW)         // 65536
#define COLS  1728                // HID*CG*K*K
#define INTERC 24
#define NTILE_W 108               // 1728/16 column tiles of kba_w
#define NTILE_B 6                 // 96/16 column tiles of kba_b
#define NTILE  (NTILE_W+NTILE_B)  // 114

typedef __attribute__((ext_vector_type(16))) __bf16 v16bf;
typedef __attribute__((ext_vector_type(8)))  float  v8f;

// ---------------------------------------------------------------------------
// Pack kba_w [32,1728] and kba_b [32,96] into WMMA B-operand lane layout (bf16).
// B 32x16 (KxN), contiguous k order chosen here (A uses the same): lanes 0-15 hold K=0..15, lanes 16-31 hold K=16..31;
// per lane 16 halves = contiguous 32B -> two clause-paired b128 loads at use.
// ---------------------------------------------------------------------------
__global__ void pack_wb(const float* __restrict__ kba_w,
                        const float* __restrict__ kba_b,
                        __bf16* __restrict__ wpack) {
  int t = blockIdx.x * 256 + threadIdx.x;
  if (t >= NTILE * 32) return;
  int tile  = t >> 5;
  int lane  = t & 31;
  int col16 = lane & 15;
  int kbase = (lane >> 4) * 16;
  __bf16* dst = wpack + ((size_t)tile * 512 + lane * 16);
  if (tile < NTILE_W) {
    int col = tile * 16 + col16;
#pragma unroll
    for (int j = 0; j < 16; ++j) dst[j] = (__bf16)kba_w[(kbase + j) * COLS + col];
  } else {
    int col = (tile - NTILE_W) * 16 + col16;
#pragma unroll
    for (int j = 0; j < 16; ++j) dst[j] = (__bf16)kba_b[(kbase + j) * HIDC + col];
  }
}

// ---------------------------------------------------------------------------
// K1: per-pixel 1x1 convs: h1 = dw1_w@x, h2 = c1_w1@x, att(skip) = c211_w@x+b.
// ---------------------------------------------------------------------------
__global__ void k1_pointwise(const float* __restrict__ x,
                             const float* __restrict__ dw1_w,
                             const float* __restrict__ c1_w1,
                             const float* __restrict__ c211_w,
                             const float* __restrict__ c211_b,
                             float* __restrict__ h1,
                             float* __restrict__ h2,
                             float* __restrict__ att) {
  __shared__ float s_dw1[HIDC * DIMC];
  __shared__ float s_c1[HIDC * DIMC];
  __shared__ float s_c211[NSET * DIMC];
  __shared__ float s_cb[NSET];
  int tid = threadIdx.x;
  for (int i = tid; i < HIDC * DIMC; i += 256) { s_dw1[i] = dw1_w[i]; s_c1[i] = c1_w1[i]; }
  for (int i = tid; i < NSET * DIMC; i += 256) s_c211[i] = c211_w[i];
  if (tid < NSET) s_cb[tid] = c211_b[tid];
  __syncthreads();
  int gp = blockIdx.x * 256 + tid;
  int b = gp / PHW, p = gp % PHW;
  float xv[DIMC];
#pragma unroll
  for (int c = 0; c < DIMC; ++c) xv[c] = x[(b * DIMC + c) * PHW + p];
  for (int oc = 0; oc < HIDC; ++oc) {
    float a1 = 0.f, a2 = 0.f;
    const float* w1 = s_dw1 + oc * DIMC;
    const float* w2 = s_c1 + oc * DIMC;
#pragma unroll
    for (int c = 0; c < DIMC; ++c) { a1 += w1[c] * xv[c]; a2 += w2[c] * xv[c]; }
    h1[(b * HIDC + oc) * PHW + p] = a1;
    h2[(b * HIDC + oc) * PHW + p] = a2;
  }
  for (int n = 0; n < NSET; ++n) {
    float a = s_cb[n];
    const float* w = s_c211 + n * DIMC;
#pragma unroll
    for (int c = 0; c < DIMC; ++c) a += w[c] * xv[c];
    att[(b * NSET + n) * PHW + p] = a;
  }
}

// ---------------------------------------------------------------------------
// K2: depthwise 3x3 (pad=1) on both branches: x1 = dw(h1,dw2_w), uf = dw(h2,c1_w2)
// ---------------------------------------------------------------------------
__global__ void k2_dw3x3(const float* __restrict__ h1,
                         const float* __restrict__ h2,
                         const float* __restrict__ dw2_w,
                         const float* __restrict__ c1_w2,
                         float* __restrict__ x1,
                         float* __restrict__ uf) {
  __shared__ float sA[HIDC * 9], sB[HIDC * 9];
  int tid = threadIdx.x;
  for (int i = tid; i < HIDC * 9; i += 256) { sA[i] = dw2_w[i]; sB[i] = c1_w2[i]; }
  __syncthreads();
  size_t t = (size_t)blockIdx.x * 256 + tid;
  int p  = (int)(t % PHW);
  int bc = (int)(t / PHW);
  int c  = bc % HIDC;
  int y = p / WWID, xc = p % WWID;
  const float* a = h1 + (size_t)bc * PHW;
  const float* g = h2 + (size_t)bc * PHW;
  float r1 = 0.f, r2 = 0.f;
#pragma unroll
  for (int ki = 0; ki < 3; ++ki)
#pragma unroll
    for (int kj = 0; kj < 3; ++kj) {
      int yy = y + ki - 1, xx = xc + kj - 1;
      if (yy < 0 || yy >= HH || xx < 0 || xx >= WWID) continue;
      int off = yy * WWID + xx;
      r1 += sA[c * 9 + ki * 3 + kj] * a[off];
      r2 += sB[c * 9 + ki * 3 + kj] * g[off];
    }
  x1[t] = r1;
  uf[t] = r2;
}

// ---------------------------------------------------------------------------
// K3: attention branch: grouped 3x3 (groups=24) -> SimpleGate -> 1x1 -> *gamma + skip
// att buffer holds c211 skip from K1; updated in place (same element, same thread).
// ---------------------------------------------------------------------------
__global__ void k3_att(const float* __restrict__ x,
                       const float* __restrict__ c2_w1, const float* __restrict__ c2_b1,
                       const float* __restrict__ c2_w2, const float* __restrict__ c2_b2,
                       const float* __restrict__ attgamma,
                       float* __restrict__ att) {
  __shared__ float s_w1[INTERC * 18], s_b1[INTERC], s_w2[NSET * 12], s_b2[NSET], s_g[NSET];
  int tid = threadIdx.x;
  for (int i = tid; i < INTERC * 18; i += 256) s_w1[i] = c2_w1[i];
  for (int i = tid; i < NSET * 12; i += 256) s_w2[i] = c2_w2[i];
  if (tid < INTERC) s_b1[tid] = c2_b1[tid];
  if (tid < NSET) { s_b2[tid] = c2_b2[tid]; s_g[tid] = attgamma[tid]; }
  __syncthreads();
  int gp = blockIdx.x * 256 + tid;
  int b = gp / PHW, p = gp % PHW;
  int y = p / WWID, xc = p % WWID;
  float tch[INTERC];
  for (int oc = 0; oc < INTERC; ++oc) {
    float acc = s_b1[oc];
#pragma unroll
    for (int ic = 0; ic < 2; ++ic) {
      const float* xp = x + (size_t)(b * DIMC + oc * 2 + ic) * PHW;
      const float* wp = s_w1 + (oc * 2 + ic) * 9;
#pragma unroll
      for (int ki = 0; ki < 3; ++ki)
#pragma unroll
        for (int kj = 0; kj < 3; ++kj) {
          int yy = y + ki - 1, xx = xc + kj - 1;
          if (yy < 0 || yy >= HH || xx < 0 || xx >= WWID) continue;
          acc += wp[ki * 3 + kj] * xp[yy * WWID + xx];
        }
    }
    tch[oc] = acc;
  }
  float sg[12];
#pragma unroll
  for (int i = 0; i < 12; ++i) sg[i] = tch[i] * tch[i + 12];
  for (int n = 0; n < NSET; ++n) {
    float acc = s_b2[n];
#pragma unroll
    for (int i = 0; i < 12; ++i) acc += s_w2[n * 12 + i] * sg[i];
    size_t idx = (size_t)(b * NSET + n) * PHW + p;
    att[idx] = acc * s_g[n] + att[idx];
  }
}

// ---------------------------------------------------------------------------
// K4: KBA core via WMMA. Wave -> 16 consecutive pixels of one image row.
// A = att[16x32] (bf16, contiguous k: lane half h holds k=16h..16h+15). Per chunk (4 groups = 144 attk cols):
//   - ALL 9 B tiles preloaded to registers first (one load clause, staggered
//     waits) then 9 WMMAs retire straight to LDS; +1 bias WMMA every other
//     chunk (16 cols -> 4KB LDS)
//   - uf halo tile (8ch x 3 x 18 = 432 floats) cooperatively staged to LDS
//   - contraction reads LDS only; float4 x1 load / comb store; 1 pool atomic/lane
// ---------------------------------------------------------------------------
__global__ void __launch_bounds__(128) k4_kba(
    const float* __restrict__ att, const float* __restrict__ uf,
    const float* __restrict__ x1,  const float* __restrict__ ga1,
    const __bf16* __restrict__ wpack,
    float* __restrict__ comb, float* __restrict__ pool) {
  __shared__ float s_attk[4][16][144];   // 36864 B
  __shared__ float s_biasc[4][16][16];   //  4096 B
  __shared__ float s_uf[4][8][3][18];    //  6912 B
  __shared__ float s_pool[HIDC];         //   384 B
  int tid  = threadIdx.x;
  int wave = tid >> 5, lane = tid & 31;
  if (tid < HIDC) s_pool[tid] = 0.f;

  int wavebase = blockIdx.x * 64 + wave * 16;   // 16 consecutive pixels, same row
  int b     = wavebase / PHW;
  int pbase = wavebase % PHW;
  int y     = pbase / WWID;
  int xbase = pbase % WWID;

  // A operand: lane<16 -> row=lane, K 0-15 ; lane>=16 -> row=lane-16, K 16-31
  int row   = lane & 15;
  int khalf = (lane >> 4) * 8;
  int pa    = pbase + row;
  v16bf av;
#pragma unroll
  for (int j = 0; j < 8; ++j) {
    av[j]     = (__bf16)att[(b * NSET + 2 * khalf + j) * PHW + pa];
    av[8 + j] = (__bf16)att[(b * NSET + 2 * khalf + 8 + j) * PHW + pa];
  }
  int mrowbase = 8 * (lane >> 4);   // D: VGPR r holds M=r (+8 for hi half-wave)
  int ncol     = lane & 15;         // D: N = lane&15

  // per-lane epilogue mapping: 4 consecutive pixels, one channel slot (go)
  int e0   = lane * 4;
  int pix0 = e0 & 15;
  int go   = e0 >> 4;          // 0..7: gl*2+o
  int gl   = go >> 1, o = go & 1;

  for (int chunk = 0; chunk < 12; ++chunk) {
    // ---- preload all 9 B tiles: loads issue as one clause, waits stagger ----
    v16bf bvs[9];
#pragma unroll
    for (int t = 0; t < 9; ++t)
      bvs[t] = *(const v16bf*)(wpack + (size_t)(chunk * 9 + t) * 512 + lane * 16);
    // ---- 9 attk WMMAs, each D retires straight to LDS ----
#pragma unroll
    for (int t = 0; t < 9; ++t) {
      v8f c = {0.f, 0.f, 0.f, 0.f, 0.f, 0.f, 0.f, 0.f};
      c = __builtin_amdgcn_wmma_f32_16x16x32_bf16(false, av, false, bvs[t],
                                                  (short)0, c, false, false);
#pragma unroll
      for (int r = 0; r < 8; ++r)
        s_attk[wave][mrowbase + r][t * 16 + ncol] = c[r];
    }
    // ---- bias tile (covers this chunk and the next; uniform branch) ----
    if ((chunk & 1) == 0) {
      v16bf bv = *(const v16bf*)(wpack + (size_t)(NTILE_W + (chunk >> 1)) * 512 + lane * 16);
      v8f c = {0.f, 0.f, 0.f, 0.f, 0.f, 0.f, 0.f, 0.f};
      c = __builtin_amdgcn_wmma_f32_16x16x32_bf16(false, av, false, bv,
                                                  (short)0, c, false, false);
#pragma unroll
      for (int r = 0; r < 8; ++r)
        s_biasc[wave][mrowbase + r][ncol] = c[r];
    }
    // ---- stage uf halo tile: channels chunk*8..+7, rows y-1..y+1, cols xbase-1..+16 ----
    for (int i = lane; i < 8 * 3 * 18; i += 32) {
      int ch  = i / 54;
      int rem = i % 54;
      int r   = rem / 18;
      int cx  = rem % 18;
      int yy  = y + r - 1;
      int xx  = xbase + cx - 1;
      float v = 0.f;
      if (yy >= 0 && yy < HH && xx >= 0 && xx < WWID)
        v = uf[(size_t)(b * HIDC + chunk * 8 + ch) * PHW + yy * WWID + xx];
      s_uf[wave][ch][r][cx] = v;
    }
    __syncthreads();

    // ---- contraction + fused epilogue (LDS-only operands) ----
    int g  = chunk * 4 + gl;
    int ch = g * 2 + o;          // global hidden channel
    size_t chbase = (size_t)(b * HIDC + ch) * PHW + pbase + pix0;
    float4 g1v = *(const float4*)(x1 + chbase);
    float ga   = ga1[ch];
    float out4[4];
    float psum = 0.f;
#pragma unroll
    for (int q = 0; q < 4; ++q) {
      int pix = pix0 + q;
      float acc = 0.f;
#pragma unroll
      for (int m = 0; m < 18; ++m) {
        int ci = m / 9, k = m % 9;
        int ki = k / 3, kj = k % 3;
        acc += s_attk[wave][pix][gl * 36 + o * 18 + m] *
               s_uf[wave][gl * 2 + ci][ki][pix + kj];
      }
      float ufc = s_uf[wave][go][1][pix + 1];
      float x2  = (acc + s_biasc[wave][pix][(chunk & 1) * 8 + go]) * ga + ufc;
      float g1  = (q == 0) ? g1v.x : (q == 1) ? g1v.y : (q == 2) ? g1v.z : g1v.w;
      float gel = 0.5f * g1 * (1.0f + erff(g1 * 0.70710678118654752f));
      float cv  = gel * x2;
      out4[q] = cv;
      psum += cv;
    }
    *(float4*)(comb + chbase) = make_float4(out4[0], out4[1], out4[2], out4[3]);
    atomicAdd(&s_pool[ch], psum);
    __syncthreads();
  }
  if (tid < HIDC) atomicAdd(&pool[b * HIDC + tid], s_pool[tid]);
}

// ---------------------------------------------------------------------------
// K5: scale[b,oc] = sca_b[oc] + sca_w @ (pool/P)
// ---------------------------------------------------------------------------
__global__ void k5_scale(const float* __restrict__ pool,
                         const float* __restrict__ sca_w,
                         const float* __restrict__ sca_b,
                         float* __restrict__ scale) {
  int i = threadIdx.x;
  if (i >= BATCH * HIDC) return;
  int b = i / HIDC, oc = i % HIDC;
  float acc = sca_b[oc];
  const float inv = 1.0f / (float)PHW;
  for (int ic = 0; ic < HIDC; ++ic)
    acc += sca_w[oc * HIDC + ic] * (pool[b * HIDC + ic] * inv);
  scale[i] = acc;
}

// ---------------------------------------------------------------------------
// K6: out = proj_w @ (comb * scale)   (scale folded into weights per block)
// ---------------------------------------------------------------------------
__global__ void k6_proj(const float* __restrict__ comb,
                        const float* __restrict__ proj_w,
                        const float* __restrict__ scale,
                        float* __restrict__ out) {
  __shared__ float s_w[DIMC * HIDC];
  int tid = threadIdx.x;
  int gp = blockIdx.x * 256 + tid;
  int b = gp / PHW, p = gp % PHW;
  for (int i = tid; i < DIMC * HIDC; i += 256)
    s_w[i] = proj_w[i] * scale[b * HIDC + (i % HIDC)];
  __syncthreads();
  float acc[DIMC];
#pragma unroll
  for (int j = 0; j < DIMC; ++j) acc[j] = 0.f;
  for (int c = 0; c < HIDC; ++c) {
    float v = comb[(size_t)(b * HIDC + c) * PHW + p];
#pragma unroll
    for (int j = 0; j < DIMC; ++j) acc[j] += s_w[j * HIDC + c] * v;
  }
#pragma unroll
  for (int j = 0; j < DIMC; ++j)
    out[(size_t)(b * DIMC + j) * PHW + p] = acc[j];
}

// ---------------------------------------------------------------------------
extern "C" void kernel_launch(void* const* d_in, const int* in_sizes, int n_in,
                              void* d_out, int out_size, void* d_ws, size_t ws_size,
                              hipStream_t stream) {
  const float* x      = (const float*)d_in[0];
  const float* dw1_w  = (const float*)d_in[1];
  const float* dw2_w  = (const float*)d_in[2];
  const float* proj_w = (const float*)d_in[3];
  const float* sca_w  = (const float*)d_in[4];
  const float* sca_b  = (const float*)d_in[5];
  const float* c1_w1  = (const float*)d_in[6];
  const float* c1_w2  = (const float*)d_in[7];
  const float* kba_w  = (const float*)d_in[8];
  const float* kba_b  = (const float*)d_in[9];
  const float* c2_w1  = (const float*)d_in[10];
  const float* c2_b1  = (const float*)d_in[11];
  const float* c2_w2  = (const float*)d_in[12];
  const float* c2_b2  = (const float*)d_in[13];
  const float* c211_w = (const float*)d_in[14];
  const float* c211_b = (const float*)d_in[15];
  const float* attg   = (const float*)d_in[16];
  const float* ga1    = (const float*)d_in[17];

  char* wsb = (char*)d_ws;
  size_t off = 0;
  auto take = [&](size_t bytes) {
    void* p = wsb + off;
    off = (off + bytes + 255) & ~(size_t)255;
    return p;
  };
  __bf16* wpack = (__bf16*)take((size_t)NTILE * 512 * sizeof(__bf16));
  float*  pool  = (float*)take((size_t)BATCH * HIDC * sizeof(float));
  float*  scale = (float*)take((size_t)BATCH * HIDC * sizeof(float));
  float*  h1    = (float*)take((size_t)NPIX * HIDC * sizeof(float));
  float*  h2    = (float*)take((size_t)NPIX * HIDC * sizeof(float));
  float*  x1b   = (float*)take((size_t)NPIX * HIDC * sizeof(float));
  float*  ufb   = (float*)take((size_t)NPIX * HIDC * sizeof(float));
  float*  attb  = (float*)take((size_t)NPIX * NSET * sizeof(float));
  float*  comb  = h1;  // h1 dead after K2; reuse for comb

  hipMemsetAsync(pool, 0, (size_t)BATCH * HIDC * sizeof(float), stream);
  pack_wb<<<(NTILE * 32 + 255) / 256, 256, 0, stream>>>(kba_w, kba_b, wpack);
  k1_pointwise<<<NPIX / 256, 256, 0, stream>>>(x, dw1_w, c1_w1, c211_w, c211_b,
                                               h1, h2, attb);
  k2_dw3x3<<<(NPIX * HIDC) / 256, 256, 0, stream>>>(h1, h2, dw2_w, c1_w2, x1b, ufb);
  k3_att<<<NPIX / 256, 256, 0, stream>>>(x, c2_w1, c2_b1, c2_w2, c2_b2, attg, attb);
  k4_kba<<<NPIX / 64, 128, 0, stream>>>(attb, ufb, x1b, ga1, wpack, comb, pool);
  k5_scale<<<1, BATCH * HIDC, 0, stream>>>(pool, sca_w, sca_b, scale);
  k6_proj<<<NPIX / 256, 256, 0, stream>>>(comb, proj_w, scale, (float*)d_out);
}
